// GroupedQueryAttention_66022237274158
// MI455X (gfx1250) — compile-verified, matrix operand order checked by reading, not run on hardware
//
#include <hip/hip_runtime.h>
#include <stdint.h>

// ---------------- problem constants ----------------
constexpr int BB   = 2;
constexpr int SS   = 2048;
constexpr int DD   = 2048;
constexpr int HH   = 32;
constexpr int KVH  = 8;
constexpr int HD   = 64;
constexpr int GRP  = HH / KVH;      // 4
constexpr int MM   = BB * SS;       // 4096 rows

typedef __attribute__((ext_vector_type(16))) __bf16 v16bf;
typedef __attribute__((ext_vector_type(8)))  float  v8f;
typedef __attribute__((ext_vector_type(4)))  unsigned int u32x4;

union FragB {
    v16bf v;
    u32x4 q[2];
};

// ---------------- elementwise cast ----------------
__global__ void cast_f32_bf16(const float* __restrict__ in, __bf16* __restrict__ out, int n) {
    int i = blockIdx.x * 256 + threadIdx.x;
    if (i < n) out[i] = (__bf16)in[i];
}

// W[Kd][Nd] f32 -> WT[Nd][Kd] bf16   (write-coalesced)
__global__ void transpose_cast(const float* __restrict__ W, __bf16* __restrict__ WT,
                               int Kd, int Nd) {
    long long idx = (long long)blockIdx.x * 256 + threadIdx.x;
    int k = (int)(idx % Kd);
    int n = (int)(idx / Kd);
    WT[(long long)n * Kd + k] = (__bf16)W[(long long)k * Nd + n];
}

// ---------------- bf16 WMMA GEMM:  C[M,N] = A[M,K] * BT[N,K]^T ----------------
// block = 256 threads (8 waves); wave tile = 32x64; block tile = 256x64.
// B operand staged through LDS with double-buffered async copies
// (global_load_async_to_lds_b128, ASYNCcnt) so the 64x32 B tile is fetched
// once per workgroup instead of once per wave.
__global__ void gemm_bf16(const __bf16* __restrict__ A, const __bf16* __restrict__ BT,
                          float* __restrict__ C, int M, int N, int K) {
    __shared__ __attribute__((aligned(16))) __bf16 bstage[2][64 * 32];   // 2 x 4KB

    const int lane  = threadIdx.x & 31;
    const int wave  = threadIdx.x >> 5;
    const int mbase = blockIdx.y * 256 + wave * 32;
    const int nbase = blockIdx.x * 64;
    const int row   = lane & 15;
    const int koff  = (lane >> 4) * 8;   // A/B fragment half-wave K offset

    // async staging roles: each thread moves one 16-byte chunk per tile
    const int tid  = threadIdx.x;
    const int srow = tid >> 2;          // 0..63 : BT row within tile
    const int scol = tid & 3;           // 0..3  : 16B chunk within 64B row
    const unsigned lds0 = (unsigned)(uintptr_t)(&bstage[0][0]);
    const unsigned ldsoff_a = lds0 + (unsigned)(srow * 64 + scol * 16);
    const unsigned ldsoff_b = ldsoff_a + 4096u;
    const __bf16* gsrc_row = BT + (long long)(nbase + srow) * K + scol * 8;

    // prologue: stage K-tile 0 into buffer 0
    {
        unsigned long long g = (unsigned long long)(uintptr_t)gsrc_row;
        asm volatile("global_load_async_to_lds_b128 %0, %1, off"
                     :: "v"(ldsoff_a), "v"(g) : "memory");
    }

    v8f acc[2][4];
    #pragma unroll
    for (int i = 0; i < 2; ++i)
        #pragma unroll
        for (int j = 0; j < 4; ++j)
            acc[i][j] = (v8f){};

    const int nk = K / 32;
    for (int it = 0; it < nk; ++it) {
        const int k0  = it * 32;
        const int cur = it & 1;

        // my chunks of the current tile have landed:
        asm volatile("s_wait_asynccnt 0x0" ::: "memory");
        // everyone's chunks have landed; everyone's reads of the other buffer are done:
        __syncthreads();
        // prefetch next tile into the other buffer (overlaps with compute below)
        if (it + 1 < nk) {
            unsigned long long g =
                (unsigned long long)(uintptr_t)(gsrc_row + (long long)(it + 1) * 32);
            asm volatile("global_load_async_to_lds_b128 %0, %1, off"
                         :: "v"(cur ? ldsoff_a : ldsoff_b), "v"(g) : "memory");
        }

        FragB a[2], b[4];
        #pragma unroll
        for (int i = 0; i < 2; ++i) {
            const __bf16* p = A + (long long)(mbase + i * 16 + row) * K + k0 + koff;
            a[i].q[0] = *reinterpret_cast<const u32x4*>(p);
            a[i].q[1] = *reinterpret_cast<const u32x4*>(p + 16);
        }
        #pragma unroll
        for (int j = 0; j < 4; ++j) {
            const __bf16* bp = &bstage[cur][(j * 16 + row) * 32 + koff];
            b[j].q[0] = *reinterpret_cast<const u32x4*>(bp);
            b[j].q[1] = *reinterpret_cast<const u32x4*>(bp + 16);
        }
        #pragma unroll
        for (int i = 0; i < 2; ++i)
            #pragma unroll
            for (int j = 0; j < 4; ++j)
                acc[i][j] = __builtin_amdgcn_wmma_f32_16x16x32_bf16(
                    false, a[i].v, false, b[j].v, (short)0, acc[i][j], false, false);
    }

    const int rb = (lane >> 4) * 8;
    #pragma unroll
    for (int i = 0; i < 2; ++i)
        #pragma unroll
        for (int j = 0; j < 4; ++j)
            #pragma unroll
            for (int r = 0; r < 8; ++r)
                C[(long long)(mbase + i * 16 + rb + r) * N + nbase + j * 16 + (lane & 15)]
                    = acc[i][j][r];
}

// ---------------- RoPE + head reorg:  Pf[B*S, heads*HD] f32 -> Pb[B,heads,S,HD] bf16 ----------------
__global__ void rope_reorg(const float* __restrict__ Pf, __bf16* __restrict__ Pb, int heads) {
    long long idx = (long long)blockIdx.x * 256 + threadIdx.x;  // over B*S*heads*HD
    int d = (int)(idx % HD);
    long long t = idx / HD;
    int h = (int)(t % heads);
    long long bs = t / heads;                  // b*S + s
    int s = (int)(bs % SS);
    int b = (int)(bs / SS);

    const float* rowp = Pf + bs * ((long long)heads * HD) + h * HD;
    float v = rowp[d];
    int i = d & 31;
    float invf = __powf(10000.0f, -(float)(2 * i) / 64.0f);
    float ang  = (float)s * invf;
    float sn, cs;
    __sincosf(ang, &sn, &cs);
    float other = (d < 32) ? -rowp[d + 32] : rowp[d - 32];
    float o = v * cs + other * sn;
    Pb[(((long long)b * heads + h) * SS + s) * HD + d] = (__bf16)o;
}

// V reorg: Vf[B*S, KV*HD] f32 -> VT[B,KV,HD,S] bf16 (write-coalesced)
__global__ void v_reorg(const float* __restrict__ Vf, __bf16* __restrict__ VT) {
    long long idx = (long long)blockIdx.x * 256 + threadIdx.x;  // over B*KV*HD*S
    int s = (int)(idx % SS);
    long long t = idx / SS;
    int d  = (int)(t % HD);
    long long t2 = t / HD;
    int kv = (int)(t2 % KVH);
    int b  = (int)(t2 / KVH);
    VT[idx] = (__bf16)Vf[((long long)b * SS + s) * (KVH * HD) + kv * HD + d];
}

// ---------------- flash attention ----------------
// grid (S/64, H, B); 128 threads = 4 waves; each wave owns one 16-row q tile.
__global__ void fa_kernel(const __bf16* __restrict__ Qb, const __bf16* __restrict__ Kb,
                          const __bf16* __restrict__ VTb, __bf16* __restrict__ attn) {
    __shared__ __attribute__((aligned(16))) __bf16 plds[4][16 * 32];

    const int lane  = threadIdx.x & 31;
    const int wave  = threadIdx.x >> 5;
    const int qtile = blockIdx.x * 4 + wave;        // 0..S/16-1
    const int h  = blockIdx.y;
    const int b  = blockIdx.z;
    const int kv = h / GRP;

    const __bf16* Qbase = Qb  + (((long long)(b * HH  + h ) * SS) + qtile * 16) * HD;
    const __bf16* Kbase = Kb  +  ((long long)(b * KVH + kv) * SS) * HD;
    const __bf16* Vbase = VTb +  ((long long)(b * KVH + kv) * HD) * SS;   // [HD][S]

    const int row  = lane & 15;
    const int hi   = lane >> 4;
    const int koff = hi * 8;

    // Q A-fragments for hd 0..31 and 32..63 (resident all loop)
    FragB qa0, qa1;
    {
        const __bf16* p = Qbase + row * HD + koff;
        qa0.q[0] = *reinterpret_cast<const u32x4*>(p);
        qa0.q[1] = *reinterpret_cast<const u32x4*>(p + 16);
        qa1.q[0] = *reinterpret_cast<const u32x4*>(p + 32);
        qa1.q[1] = *reinterpret_cast<const u32x4*>(p + 48);
    }

    v8f acc[4];
    #pragma unroll
    for (int t = 0; t < 4; ++t) acc[t] = (v8f){};
    float m[8], lsum[8];
    #pragma unroll
    for (int r = 0; r < 8; ++r) { m[r] = -1e30f; lsum[r] = 0.0f; }

    const float sm_scale = 0.125f;   // 1/sqrt(64)

    for (int kb = 0; kb < SS; kb += 32) {
        // prefetch next key block while we compute on this one
        if (kb + 32 < SS)
            __builtin_prefetch(Kbase + (long long)(kb + 32 + row) * HD, 0, 0);

        // ---- scores: two 16x16 tiles (keys kb..+15, kb+16..+31), hd = 64 via 2 chained WMMAs
        v8f c0 = (v8f){}, c1 = (v8f){};
        {
            FragB k0a, k0b;
            const __bf16* kp = Kbase + (long long)(kb + row) * HD + koff;
            k0a.q[0] = *reinterpret_cast<const u32x4*>(kp);
            k0a.q[1] = *reinterpret_cast<const u32x4*>(kp + 16);
            k0b.q[0] = *reinterpret_cast<const u32x4*>(kp + 32);
            k0b.q[1] = *reinterpret_cast<const u32x4*>(kp + 48);
            c0 = __builtin_amdgcn_wmma_f32_16x16x32_bf16(false, qa0.v, false, k0a.v, (short)0, c0, false, false);
            c0 = __builtin_amdgcn_wmma_f32_16x16x32_bf16(false, qa1.v, false, k0b.v, (short)0, c0, false, false);

            FragB k1a, k1b;
            const __bf16* kp2 = kp + 16 * HD;
            k1a.q[0] = *reinterpret_cast<const u32x4*>(kp2);
            k1a.q[1] = *reinterpret_cast<const u32x4*>(kp2 + 16);
            k1b.q[0] = *reinterpret_cast<const u32x4*>(kp2 + 32);
            k1b.q[1] = *reinterpret_cast<const u32x4*>(kp2 + 48);
            c1 = __builtin_amdgcn_wmma_f32_16x16x32_bf16(false, qa0.v, false, k1a.v, (short)0, c1, false, false);
            c1 = __builtin_amdgcn_wmma_f32_16x16x32_bf16(false, qa1.v, false, k1b.v, (short)0, c1, false, false);
        }

        // ---- online softmax (row = rb+r lives across one 16-lane half)
        float p0[8], p1[8];
        #pragma unroll
        for (int r = 0; r < 8; ++r) {
            float s0 = c0[r] * sm_scale;
            float s1 = c1[r] * sm_scale;
            float t = fmaxf(s0, s1);
            t = fmaxf(t, __shfl_xor(t, 1, 32));
            t = fmaxf(t, __shfl_xor(t, 2, 32));
            t = fmaxf(t, __shfl_xor(t, 4, 32));
            t = fmaxf(t, __shfl_xor(t, 8, 32));
            float mn = fmaxf(m[r], t);
            float f  = __expf(m[r] - mn);
            p0[r] = __expf(s0 - mn);
            p1[r] = __expf(s1 - mn);
            float rs = p0[r] + p1[r];
            rs += __shfl_xor(rs, 1, 32);
            rs += __shfl_xor(rs, 2, 32);
            rs += __shfl_xor(rs, 4, 32);
            rs += __shfl_xor(rs, 8, 32);
            lsum[r] = lsum[r] * f + rs;
            m[r] = mn;
            acc[0][r] *= f; acc[1][r] *= f; acc[2][r] *= f; acc[3][r] *= f;
        }

        // ---- P (C-layout) -> LDS -> A-fragment layout
        __bf16* pl = plds[wave];
        const int rb = hi * 8;
        #pragma unroll
        for (int r = 0; r < 8; ++r) {
            pl[(rb + r) * 32 + row]      = (__bf16)p0[r];
            pl[(rb + r) * 32 + 16 + row] = (__bf16)p1[r];
        }
        asm volatile("s_wait_dscnt 0" ::: "memory");

        FragB pa;
        {
            const __bf16* pr = pl + row * 32 + koff;
            pa.q[0] = *reinterpret_cast<const u32x4*>(pr);
            pa.q[1] = *reinterpret_cast<const u32x4*>(pr + 16);
        }

        // ---- P @ V : 4 hd tiles, B operand rows from VT [hd][S]
        #pragma unroll
        for (int t = 0; t < 4; ++t) {
            FragB vb;
            const __bf16* vp = Vbase + (long long)(t * 16 + row) * SS + kb + koff;
            vb.q[0] = *reinterpret_cast<const u32x4*>(vp);
            vb.q[1] = *reinterpret_cast<const u32x4*>(vp + 16);
            acc[t] = __builtin_amdgcn_wmma_f32_16x16x32_bf16(
                false, pa.v, false, vb.v, (short)0, acc[t], false, false);
        }
    }

    // ---- finalize: acc/l -> attn[b, q, h*64 + d] bf16
    __bf16* obase = attn + ((long long)b * SS + qtile * 16) * (HH * HD) + h * HD;
    #pragma unroll
    for (int r = 0; r < 8; ++r) {
        float inv = 1.0f / lsum[r];
        __bf16* op = obase + (long long)(hi * 8 + r) * (HH * HD);
        op[0 * 16 + row] = (__bf16)(acc[0][r] * inv);
        op[1 * 16 + row] = (__bf16)(acc[1][r] * inv);
        op[2 * 16 + row] = (__bf16)(acc[2][r] * inv);
        op[3 * 16 + row] = (__bf16)(acc[3][r] * inv);
    }
}

// ---------------- host launch ----------------
extern "C" void kernel_launch(void* const* d_in, const int* in_sizes, int n_in,
                              void* d_out, int out_size, void* d_ws, size_t ws_size,
                              hipStream_t stream) {
    const float* x  = (const float*)d_in[0];
    const float* Wq = (const float*)d_in[1];
    const float* Wk = (const float*)d_in[2];
    const float* Wv = (const float*)d_in[3];
    const float* Wo = (const float*)d_in[4];
    float* out = (float*)d_out;

    char* ws = (char*)d_ws;
    // static carve (bytes)
    __bf16* xb   = (__bf16*)(ws);                                  // 16 MB
    __bf16* WqT  = (__bf16*)(ws + 16777216);                       //  8 MB
    __bf16* WkT  = (__bf16*)(ws + 16777216 + 8388608);             //  2 MB
    __bf16* WvT  = (__bf16*)(ws + 16777216 + 8388608 + 2097152);   //  2 MB
    __bf16* WoT  = (__bf16*)(ws + 16777216 + 8388608 + 4194304);   //  8 MB
    char*   base = ws + 16777216 + 8388608 + 4194304 + 8388608;    // = ws + 37748736
    float*  Qf   = (float*)(base);                                 // 32 MB
    float*  Kf   = (float*)(base + 33554432);                      //  8 MB
    float*  Vf   = (float*)(base + 33554432 + 8388608);            //  8 MB
    __bf16* Qb   = (__bf16*)(base + 33554432 + 16777216);          // 16 MB
    __bf16* Kb   = (__bf16*)(base + 33554432 + 16777216 + 16777216);        // 4 MB
    __bf16* VTb  = (__bf16*)(base + 33554432 + 16777216 + 16777216 + 4194304); // 4 MB
    __bf16* attn = (__bf16*)(base + 33554432);                     // aliases Kf+Vf (dead by then), 16 MB

    // 1) cast x
    cast_f32_bf16<<<(MM * DD) / 256, 256, 0, stream>>>(x, xb, MM * DD);
    // 2) transpose+cast weights
    transpose_cast<<<(DD * (HH  * HD)) / 256, 256, 0, stream>>>(Wq, WqT, DD, HH  * HD);
    transpose_cast<<<(DD * (KVH * HD)) / 256, 256, 0, stream>>>(Wk, WkT, DD, KVH * HD);
    transpose_cast<<<(DD * (KVH * HD)) / 256, 256, 0, stream>>>(Wv, WvT, DD, KVH * HD);
    transpose_cast<<<((HH * HD) * DD) / 256, 256, 0, stream>>>(Wo, WoT, HH * HD, DD);

    // 3) QKV projections (WMMA bf16 -> f32)
    gemm_bf16<<<dim3((HH  * HD) / 64, MM / 256), 256, 0, stream>>>(xb, WqT, Qf, MM, HH  * HD, DD);
    gemm_bf16<<<dim3((KVH * HD) / 64, MM / 256), 256, 0, stream>>>(xb, WkT, Kf, MM, KVH * HD, DD);
    gemm_bf16<<<dim3((KVH * HD) / 64, MM / 256), 256, 0, stream>>>(xb, WvT, Vf, MM, KVH * HD, DD);

    // 4) RoPE + reorg
    rope_reorg<<<(MM * HH  * HD) / 256, 256, 0, stream>>>(Qf, Qb, HH);
    rope_reorg<<<(MM * KVH * HD) / 256, 256, 0, stream>>>(Kf, Kb, KVH);
    v_reorg   <<<(MM * KVH * HD) / 256, 256, 0, stream>>>(Vf, VTb);

    // 5) flash attention
    fa_kernel<<<dim3(SS / 64, HH, BB), 128, 0, stream>>>(Qb, Kb, VTb, attn);

    // 6) output projection -> f32 d_out
    gemm_bf16<<<dim3(DD / 64, MM / 256), 256, 0, stream>>>(attn, WoT, out, MM, DD, DD);
}
